// VectorQuantizer_68582037783144
// MI455X (gfx1250) — matrix operand order repaired and checked by reading, not compiled, not run on hardware
//
#include <hip/hip_runtime.h>

typedef __attribute__((ext_vector_type(16))) _Float16 v16h;
typedef __attribute__((ext_vector_type(8)))  float    v8f;
typedef unsigned int u32x4 __attribute__((ext_vector_type(4)));
typedef int          i32x8 __attribute__((ext_vector_type(8)));
typedef int          i32x4 __attribute__((ext_vector_type(4)));

#define D_DIM   256
#define K_CB    4096
#define S_LEN   2048
#define N_ROWS  (32 * 2048)
#define CB_SCALE 4096.0f                         // lift codebook out of f16 subnormals

#define ROW_STRIDE_H 264                         // padded halfs per codeword row in LDS
#define ROW_STRIDE_B (ROW_STRIDE_H * 2)          // 528 B: conflict-free b128 reads
#define GROUP_TILES  8                           // 8 codeword-tiles of 16 per group
#define GROUP_CW     (GROUP_TILES * 16)          // 128 codewords / group
#define NGROUPS      (K_CB / GROUP_CW)           // 32
#define M_TILES      4                           // 4x M register blocking per wave
#define WAVES        4
#define BLOCK_THREADS (WAVES * 32)
#define ROWS_PER_WAVE (M_TILES * 16)             // 64
#define ROWS_PER_WG   (WAVES * ROWS_PER_WAVE)    // 256
#define BUF_BYTES    (GROUP_CW * ROW_STRIDE_B)   // 67584 (one buffer)
#define LDS_B_BYTES  (2 * BUF_BYTES)             // 135168 (double-buffered)
#define LDS_IDX_OFF  LDS_B_BYTES
#define LDS_RED_OFF  (LDS_IDX_OFF + ROWS_PER_WG * 4)
#define LDS_TOTAL    (LDS_RED_OFF + WAVES * 4)   // 136208 bytes

// Detect the CDNA5 Tensor Data Mover builtin.
#if defined(__AMDGCN__) && defined(__has_builtin)
# if __has_builtin(__builtin_amdgcn_tensor_load_to_lds) && \
     __has_builtin(__builtin_amdgcn_s_wait_tensorcnt)
#  define VQ_HAS_TDM 1
# endif
#endif
#ifndef VQ_HAS_TDM
# define VQ_HAS_TDM 0
#endif

// ---------------------------------------------------------------------------
// Prep: codebook f32 -> f16*4096 (into ws), scaled half-norms, zero the loss.
// ---------------------------------------------------------------------------
__global__ void __launch_bounds__(32)
vq_prep(const float* __restrict__ cb, _Float16* __restrict__ cb16,
        float* __restrict__ halfnorm, float* __restrict__ loss_out)
{
    const int c    = blockIdx.x;
    const int lane = threadIdx.x;
    const float4* src = (const float4*)(cb + (size_t)c * D_DIM);
    const float4 v0 = src[lane * 2 + 0];
    const float4 v1 = src[lane * 2 + 1];
    union { _Float16 h[8]; uint4 q; } u;
    u.h[0] = (_Float16)(v0.x * CB_SCALE); u.h[1] = (_Float16)(v0.y * CB_SCALE);
    u.h[2] = (_Float16)(v0.z * CB_SCALE); u.h[3] = (_Float16)(v0.w * CB_SCALE);
    u.h[4] = (_Float16)(v1.x * CB_SCALE); u.h[5] = (_Float16)(v1.y * CB_SCALE);
    u.h[6] = (_Float16)(v1.z * CB_SCALE); u.h[7] = (_Float16)(v1.w * CB_SCALE);
    ((uint4*)cb16)[(size_t)c * 32 + lane] = u.q;

    float s = v0.x * v0.x + v0.y * v0.y + v0.z * v0.z + v0.w * v0.w
            + v1.x * v1.x + v1.y * v1.y + v1.z * v1.z + v1.w * v1.w;
#pragma unroll
    for (int off = 16; off >= 1; off >>= 1) s += __shfl_xor(s, off, 32);
    if (lane == 0) halfnorm[c] = (0.5f * CB_SCALE) * s;   // 0.5*||e||^2 * scale
    if (c == 0 && lane == 0) *loss_out = 0.0f;
}

#if VQ_HAS_TDM
// ---------------------------------------------------------------------------
// One TDM descriptor: 2D tile 256 halfs x GROUP_CW rows, global -> LDS,
// hardware-padded rows: 512B data + 16B pad (pad_interval=128dw, amount=4dw)
// -> 528B LDS row stride (matches compute-side indexing).
// clang-23 / therock form: 6 args (g0 v4u, g1 v8i, g2 v4i, g3 v4i, g4 v8i, cpol).
// Groups 2/3 unused for 2D tensors; VADDR4 unused -> zero-filled.
// ---------------------------------------------------------------------------
__device__ __forceinline__ void tdm_load_group(const _Float16* gsrc,
                                               unsigned lds_byte_addr)
{
    const unsigned long long ga = (unsigned long long)(uintptr_t)gsrc;
    u32x4 g0;
    g0[0] = 1u;                                       // count=1, no gather
    g0[1] = lds_byte_addr;                            // lds_addr [63:32]
    g0[2] = (unsigned)(ga & 0xffffffffu);             // global_addr low
    g0[3] = (unsigned)((ga >> 32) & 0x01ffffffu)      // global_addr [56:32]
          | (2u << 30);                               // type=2 (image)
    i32x8 g1;
    g1[0] = (1 << 16)                                 // data_size=1 -> 2 bytes
          | (1 << 20)                                 // pad_enable
          | (6 << 22)                                 // pad_interval: 128 DWORDs
          | (3 << 25);                                // pad_amount: 4 DWORDs
    g1[1] = (int)((D_DIM & 0xffff) << 16);            // tensor_dim0 lo16 [63:48]
    g1[2] = (int)((GROUP_CW & 0xffff) << 16);         // tensor_dim0 hi=0, tensor_dim1 lo16
    g1[3] = (int)((D_DIM & 0xffff) << 16);            // tensor_dim1 hi=0, tile_dim0 [127:112]
    g1[4] = (int)(GROUP_CW & 0xffff);                 // tile_dim1 [143:128], tile_dim2=0
    g1[5] = D_DIM;                                    // tensor_dim0_stride lo32
    g1[6] = 0;                                        // stride hi, tensor_dim1_stride lo
    g1[7] = 0;
    const i32x4 z4 = {0, 0, 0, 0};
    const i32x8 z8 = {0, 0, 0, 0, 0, 0, 0, 0};
    __builtin_amdgcn_tensor_load_to_lds(g0, g1, z4, z4, z8, 0);
}
#endif

// ---------------------------------------------------------------------------
// Main: WMMA distance GEMM + argmin + gather + straight-through out + loss.
// ---------------------------------------------------------------------------
__global__ void __launch_bounds__(BLOCK_THREADS)
vq_main(const float* __restrict__ in, const float* __restrict__ cb,
        const _Float16* __restrict__ cb16, const float* __restrict__ halfnorm,
        float* __restrict__ out, float* __restrict__ loss)
{
    extern __shared__ char smem[];
    const int tid  = threadIdx.x;
    const int lane = tid & 31;
    const int wave = tid >> 5;

    // ---- Load A fragments: ISA 16-bit A 16x32 layout, f32 -> f16 ----
    const int m   = lane & 15;
    const int kb8 = (lane < 16) ? 0 : 8;
    const int rowbase = blockIdx.x * ROWS_PER_WG + wave * ROWS_PER_WAVE;
    const int bimg = rowbase >> 11;
    const int s0   = rowbase & (S_LEN - 1);
    const float* inB = in + (size_t)bimg * D_DIM * S_LEN;

    v16h afrag[M_TILES][8];
#pragma unroll
    for (int mt = 0; mt < M_TILES; ++mt) {
        const int srow = s0 + mt * 16 + m;
#pragma unroll
        for (int kc = 0; kc < 8; ++kc) {
#pragma unroll
            for (int v = 0; v < 8; ++v) {
                const int d = kc * 32 + kb8 + ((v < 4) ? (v * 2) : (16 + (v - 4) * 2));
                const float x0 = inB[(size_t)d * S_LEN + srow];
                const float x1 = inB[(size_t)(d + 1) * S_LEN + srow];
                afrag[mt][kc][2 * v]     = (_Float16)x0;
                afrag[mt][kc][2 * v + 1] = (_Float16)x1;
            }
        }
    }

    float bestv[M_TILES][8];
    int   besti[M_TILES][8];
#pragma unroll
    for (int mt = 0; mt < M_TILES; ++mt)
#pragma unroll
        for (int r = 0; r < 8; ++r) { bestv[mt][r] = -3.0e38f; besti[mt][r] = 0; }

    const int ncol   = lane & 15;
    const int kb16x2 = (lane < 16) ? 0 : 16;
    const v8f zero8 = {0.f, 0.f, 0.f, 0.f, 0.f, 0.f, 0.f, 0.f};
#if !VQ_HAS_TDM
    const uint4* __restrict__ wsrc = (const uint4*)cb16;
#endif

#if VQ_HAS_TDM
    const unsigned lds_base_addr = (unsigned)(uintptr_t)smem;
    if (wave == 0) tdm_load_group(cb16, lds_base_addr);        // prefetch group 0
#endif

#pragma unroll 1
    for (int g = 0; g < NGROUPS; ++g) {
#if VQ_HAS_TDM
        if (wave == 0) {
            if (g + 1 < NGROUPS) {                             // prefetch next group
                tdm_load_group(cb16 + (size_t)(g + 1) * GROUP_CW * D_DIM,
                               lds_base_addr + ((g + 1) & 1) * BUF_BYTES);
                __builtin_amdgcn_s_wait_tensorcnt(1);          // group g landed
            } else {
                __builtin_amdgcn_s_wait_tensorcnt(0);
            }
        }
        __syncthreads();
#else
        char* dstB = smem + (g & 1) * BUF_BYTES;
        const int gb = g * GROUP_CW * (D_DIM / 8);
#pragma unroll 4
        for (int j = 0; j < (GROUP_CW * 32) / BLOCK_THREADS; ++j) {
            const int c = j * BLOCK_THREADS + tid;
            const int r = c >> 5, o = c & 31;
            *(uint4*)(dstB + (size_t)r * ROW_STRIDE_B + o * 16) = wsrc[gb + r * 32 + o];
        }
        __syncthreads();
#endif
        const char* sB = smem + (g & 1) * BUF_BYTES;

#pragma unroll 1
        for (int t = 0; t < GROUP_TILES; ++t) {
            const int nglob = g * GROUP_CW + t * 16 + ncol;
            const float hn = halfnorm[nglob];
            const char* brow = sB + (size_t)(t * 16 + ncol) * ROW_STRIDE_B + kb16x2;

            v8f acc[M_TILES];
#pragma unroll
            for (int mt = 0; mt < M_TILES; ++mt) acc[mt] = zero8;

#pragma unroll
            for (int kc = 0; kc < 8; ++kc) {
                union { uint4 q[2]; v16h h; } bu;
                bu.q[0] = *(const uint4*)(brow + kc * 64);
                bu.q[1] = *(const uint4*)(brow + kc * 64 + 32);
#pragma unroll
                for (int mt = 0; mt < M_TILES; ++mt)
                    acc[mt] = __builtin_amdgcn_wmma_f32_16x16x32_f16(
                        false, afrag[mt][kc], false, bu.h, (short)0, acc[mt],
                        false, false);
            }
            // score = scale*(x.e - 0.5||e||^2); argmax == argmin distance
#pragma unroll
            for (int mt = 0; mt < M_TILES; ++mt)
#pragma unroll
                for (int r = 0; r < 8; ++r) {
                    const float sc = acc[mt][r] - hn;
                    if (sc > bestv[mt][r]) { bestv[mt][r] = sc; besti[mt][r] = nglob; }
                }
        }
        __syncthreads();   // reads done before buffer (g+1 path) is overwritten
    }

    // ---- cross-lane argmax (16 lanes per row), lowest-index tie-break ----
    int* ls_idx = (int*)(smem + LDS_IDX_OFF);
#pragma unroll
    for (int mt = 0; mt < M_TILES; ++mt) {
#pragma unroll
        for (int r = 0; r < 8; ++r) {
            float bv = bestv[mt][r];
            int   bi = besti[mt][r];
#pragma unroll
            for (int off = 1; off < 16; off <<= 1) {
                const float ov = __shfl_xor(bv, off, 32);
                const int   oi = __shfl_xor(bi, off, 32);
                if (ov > bv || (ov == bv && oi < bi)) { bv = ov; bi = oi; }
            }
            if (lane == 0)  ls_idx[wave * ROWS_PER_WAVE + mt * 16 + r] = bi;
            if (lane == 16) ls_idx[wave * ROWS_PER_WAVE + mt * 16 + 8 + r] = bi;
        }
    }
    __syncthreads();

    // ---- gather f32 codebook, straight-through output, loss partial ----
    const int row0 = blockIdx.x * ROWS_PER_WG;
    const int bo = row0 >> 11;
    const int so = row0 & (S_LEN - 1);
    const float* inO  = in  + (size_t)bo * D_DIM * S_LEN + so;
    float*       outO = out + (size_t)bo * D_DIM * S_LEN + so;

    const int dcol = (tid >= 64) ? 1 : 0;
    const int rl   = (tid & 63) * 4;
    const float* c0 = cb + (size_t)ls_idx[rl + 0] * D_DIM;
    const float* c1 = cb + (size_t)ls_idx[rl + 1] * D_DIM;
    const float* c2 = cb + (size_t)ls_idx[rl + 2] * D_DIM;
    const float* c3 = cb + (size_t)ls_idx[rl + 3] * D_DIM;

    float lsum = 0.f;
#pragma unroll 2
    for (int dblk = 0; dblk < D_DIM / 2; ++dblk) {
        const int d = dblk * 2 + dcol;
        const size_t off = (size_t)d * S_LEN + rl;
        const float4 x = *(const float4*)(inO + off);
        const float q0 = c0[d], q1 = c1[d], q2 = c2[d], q3 = c3[d];
        const float d0 = q0 - x.x, d1 = q1 - x.y, d2 = q2 - x.z, d3 = q3 - x.w;
        float4 o;
        o.x = x.x + d0; o.y = x.y + d1; o.z = x.z + d2; o.w = x.w + d3;  // STE rounding
        *(float4*)(outO + off) = o;
        lsum += d0 * d0 + d1 * d1 + d2 * d2 + d3 * d3;
    }
#pragma unroll
    for (int off = 16; off >= 1; off >>= 1) lsum += __shfl_xor(lsum, off, 32);
    float* lred = (float*)(smem + LDS_RED_OFF);
    if (lane == 0) lred[wave] = lsum;
    __syncthreads();
    if (tid == 0) {
        const float tot = lred[0] + lred[1] + lred[2] + lred[3];
        atomicAdd(loss, tot * (1.25f / 16777216.0f));   // (0.25 + 1.0) * mean
    }
}

// ---------------------------------------------------------------------------
extern "C" void kernel_launch(void* const* d_in, const int* in_sizes, int n_in,
                              void* d_out, int out_size, void* d_ws, size_t ws_size,
                              hipStream_t stream)
{
    const float* d_inputs = (const float*)d_in[0];   // [32, 256, 2048] f32
    const float* d_cb     = (const float*)d_in[1];   // [4096, 256] f32
    float* out  = (float*)d_out;                      // 16777216 quantized + 1 loss
    _Float16* cb16 = (_Float16*)d_ws;                 // 2 MB scaled f16 codebook
    float* hnorm = (float*)((char*)d_ws + (size_t)K_CB * D_DIM * 2);  // 16 KB
    float* loss = out + (size_t)N_ROWS * D_DIM;

    (void)in_sizes; (void)n_in; (void)out_size; (void)ws_size;

    (void)hipFuncSetAttribute((const void*)vq_main,
                              hipFuncAttributeMaxDynamicSharedMemorySize, LDS_TOTAL);

    vq_prep<<<K_CB, 32, 0, stream>>>(d_cb, cb16, hnorm, loss);
    vq_main<<<N_ROWS / ROWS_PER_WG, BLOCK_THREADS, LDS_TOTAL, stream>>>(
        d_inputs, d_cb, cb16, hnorm, out, loss);
}
